// BayesBVPGenerator_15152644621024
// MI455X (gfx1250) — matrix operand order repaired and checked by reading, not compiled, not run on hardware
//
#include <hip/hip_runtime.h>
#include <hip/hip_bf16.h>
#include <math.h>

// ---------------------------------------------------------------------------
// Types for WMMA (gfx1250, wave32): v_wmma_f32_16x16x32_bf16
// ---------------------------------------------------------------------------
typedef __attribute__((ext_vector_type(16))) __bf16 v16bf;
typedef __attribute__((ext_vector_type(8)))  __bf16 v8bf;
typedef __attribute__((ext_vector_type(8)))  float  v8f;

__device__ __forceinline__ v8f bfwmma(v16bf a, v16bf b, v8f c) {
    return __builtin_amdgcn_wmma_f32_16x16x32_bf16(false, a, false, b,
                                                   (short)0, c, false, false);
}

__device__ __forceinline__ __bf16 f2bf(float f) {
    unsigned u = __builtin_bit_cast(unsigned, f);
    unsigned r = (u + 0x7FFFu + ((u >> 16) & 1u)) >> 16;
    return __builtin_bit_cast(__bf16, (unsigned short)r);
}

__device__ __forceinline__ float sigmoidf_(float x) {
    return 1.0f / (1.0f + __expf(-x));
}
// branchless tanh: (e^2x - 1)/(e^2x + 1), clamped so e^2x stays finite
__device__ __forceinline__ float tanhf_fast(float x) {
    x = fminf(fmaxf(x, -15.f), 15.f);
    float e = __expf(2.0f * x);
    return (e - 1.0f) / (e + 1.0f);
}

// A-fragment loader: 16x32 bf16 A tile, per-lane two contiguous 16B chunks.
// rowBase points at row (M = lane&15) of a [.,512] bf16 row-major matrix.
__device__ __forceinline__ v16bf loadA16(const __bf16* rowBase, int kc, int khalf) {
    v8bf lo = *(const v8bf*)(rowBase + kc + 16 * khalf);
    v8bf hi = *(const v8bf*)(rowBase + kc + 16 * khalf + 8);
    v16bf a;
#pragma unroll
    for (int i = 0; i < 8; ++i) { a[i] = lo[i]; a[i + 8] = hi[i]; }
    return a;
}

// B-fragment loader from LDS (rows padded to 520 bf16 => bank-conflict-free)
__device__ __forceinline__ v16bf loadB_lds(const __bf16* p) {
    v8bf lo = *(const v8bf*)(p);
    v8bf hi = *(const v8bf*)(p + 8);
    v16bf b;
#pragma unroll
    for (int i = 0; i < 8; ++i) { b[i] = lo[i]; b[i + 8] = hi[i]; }
    return b;
}

#define LDS_ROW   520                      // 512 + 8 pad (bf16 elements)
#define LDS_MAT   (64 * LDS_ROW)           // one weight slice: 64 rows
#define LDS_BYTES (3 * LDS_MAT * 2)        // 199680 B (< 320KB WGP LDS)

// ---------------------------------------------------------------------------
// Workspace layout (byte offsets, all 256-aligned)
// ---------------------------------------------------------------------------
#define WS_CNT   ((size_t)0)                              // unsigned[1] barrier counter
#define WS_PBUF  ((size_t)256)                            // bf16[2][64][512] layer0 h double-buffer
#define WS_QBUF  (WS_PBUF + (size_t)2*64*512*2)           // bf16[2][64][512] layer1 h double-buffer
#define WS_XIN2  (WS_QBUF + (size_t)2*64*512*2)           // f32 [64][1024] concat(h, le)
#define WS_GX0   (WS_XIN2 + (size_t)64*1024*4)            // f32 [64][2048] time-invariant gates
#define WS_WHH0  (WS_GX0  + (size_t)64*2048*4)            // bf16[2048][512]
#define WS_WIH1  (WS_WHH0 + (size_t)2048*512*2)           // bf16[2048][512]
#define WS_WHH1  (WS_WIH1 + (size_t)2048*512*2)           // bf16[2048][512]
#define WS_W1S   (WS_WHH1 + (size_t)2048*512*2)           // bf16[256][512] sig_w1
#define WS_B1S   (WS_W1S  + (size_t)256*512*2)            // f32 [2048] b_ih1+b_hh1
#define WS_CH    (WS_B1S  + (size_t)2048*4)               // bf16[64][1024][512] layer1 outputs
#define WS_HAVG  (WS_CH   + (size_t)64*1024*512*2)        // f32 [64][512]
#define WS_YOSC  (WS_HAVG + (size_t)64*512*4)             // f32 [64][256]
#define WS_OSCP  (WS_YOSC + (size_t)64*256*4)             // f32 [64][4] freq/amp/phase
#define WS_BASE  (WS_OSCP + (size_t)64*4*4)               // f32 [64][1024]

// ---------------------------------------------------------------------------
// K0: zero (barrier counter + both hidden-state double buffers)
// ---------------------------------------------------------------------------
__global__ void zero_kernel(unsigned* p, int n) {
    int i = blockIdx.x * blockDim.x + threadIdx.x;
    if (i < n) p[i] = 0u;
}

// ---------------------------------------------------------------------------
// K1a: le = emb[labels]; h = lrelu(ln(concat(z,le)@np_w^T + np_b)); xin2=[h,le]
// ---------------------------------------------------------------------------
__global__ __launch_bounds__(256) void prep_kernel(
    const float* __restrict__ z, const int* __restrict__ labels,
    const float* __restrict__ emb, const float* __restrict__ npw,
    const float* __restrict__ npb, const float* __restrict__ npg,
    const float* __restrict__ npbeta, float* __restrict__ xin2) {
    __shared__ float xc[640];
    __shared__ float pre[512];
    __shared__ float red[2];
    const int b = blockIdx.x, tid = threadIdx.x;
    const int lb = labels[b];
    for (int i = tid; i < 128; i += 256) xc[i] = z[b * 128 + i];
    for (int i = tid; i < 512; i += 256) xc[128 + i] = emb[lb * 512 + i];
    __syncthreads();
    for (int j = tid; j < 512; j += 256) {
        const float* wr = npw + (size_t)j * 640;
        float s = npb[j];
        for (int k = 0; k < 640; ++k) s += xc[k] * wr[k];
        pre[j] = s;
    }
    __syncthreads();
    if (tid == 0) {
        float m = 0.f;
        for (int j = 0; j < 512; ++j) m += pre[j];
        m *= (1.f / 512.f);
        float v = 0.f;
        for (int j = 0; j < 512; ++j) { float d = pre[j] - m; v += d * d; }
        v *= (1.f / 512.f);
        red[0] = m; red[1] = rsqrtf(v + 1e-5f);
    }
    __syncthreads();
    const float m = red[0], inv = red[1];
    for (int j = tid; j < 512; j += 256) {
        float xn = (pre[j] - m) * inv * npg[j] + npbeta[j];
        xn = xn >= 0.f ? xn : 0.2f * xn;
        xin2[(size_t)b * 1024 + j] = xn;
        xin2[(size_t)b * 1024 + 512 + j] = xc[128 + j];
    }
}

// ---------------------------------------------------------------------------
// K1b: gx0[b][n] = xin2[b] . w_ih0[n] + b_ih0[n] + b_hh0[n]   (time-invariant)
// ---------------------------------------------------------------------------
__global__ void gx0_kernel(const float* __restrict__ xin2,
                           const float* __restrict__ wih0,
                           const float* __restrict__ bih0,
                           const float* __restrict__ bhh0,
                           float* __restrict__ gx0) {
    const int idx = blockIdx.x * blockDim.x + threadIdx.x;   // 131072
    const int b = idx >> 11, n = idx & 2047;
    const float* xr = xin2 + (size_t)b * 1024;
    const float* wr = wih0 + (size_t)n * 1024;
    float s = bih0[n] + bhh0[n];
    for (int k = 0; k < 1024; ++k) s += xr[k] * wr[k];
    gx0[idx] = s;
}

// ---------------------------------------------------------------------------
// K1c: f32 -> bf16 repack;  K1d: bias sum
// ---------------------------------------------------------------------------
__global__ void cvt_kernel(const float* __restrict__ s, __bf16* __restrict__ d, int n) {
    int i = blockIdx.x * blockDim.x + threadIdx.x;
    if (i < n) d[i] = f2bf(s[i]);
}
__global__ void bsum_kernel(const float* __restrict__ a, const float* __restrict__ b,
                            float* __restrict__ o, int n) {
    int i = blockIdx.x * blockDim.x + threadIdx.x;
    if (i < n) o[i] = a[i] + b[i];
}

// ---------------------------------------------------------------------------
// Grid barrier (32 cooperative WGs, monotone counter zeroed per launch)
// ---------------------------------------------------------------------------
__device__ __forceinline__ void grid_barrier(unsigned* cnt, unsigned target) {
    __syncthreads();
    __threadfence();                 // release (global_wb + waits)
    if (threadIdx.x == 0) {
        atomicAdd(cnt, 1u);
        while (atomicAdd(cnt, 0u) < target) __builtin_amdgcn_s_sleep(1);
    }
    __syncthreads();
    __threadfence();                 // acquire (global_inv)
}

// ---------------------------------------------------------------------------
// K2: fused 2-layer LSTM scan. 32 WGs x 128 threads (4 waves).
//  WG g owns hidden columns [16g, 16g+16). Its full gate-weight slices for
//  Whh0 / Wih1 / Whh1 (3 x 64 rows x 512 = 192KB bf16) are loaded to LDS once
//  and feed all WMMA B-fragments for 1024 steps (zero steady-state L2 weight
//  traffic; rows padded to 520 elems for conflict-free ds_load_b128).
//  Wave w = batch tile mt (rows 16w..16w+15). Each wave keeps gx0, biases,
//  c0/c1 and the h-avg accumulator in registers for the whole scan.
// ---------------------------------------------------------------------------
__global__ __launch_bounds__(128, 1) void lstm_scan_kernel(
    const float* __restrict__ gx0,     // [64][2048]
    const __bf16* __restrict__ Whh0,   // [2048][512]
    const __bf16* __restrict__ Wih1,   // [2048][512]
    const __bf16* __restrict__ Whh1,   // [2048][512]
    const float* __restrict__ b1s,     // [2048]
    __bf16* __restrict__ Pbuf,         // [2][64][512]
    __bf16* __restrict__ Qbuf,         // [2][64][512]
    __bf16* __restrict__ ch,           // [64][1024][512]
    float* __restrict__ havg,          // [64][512]
    unsigned* __restrict__ cnt) {
    extern __shared__ __bf16 smem[];       // [3][64][520]
    const int g32   = blockIdx.x;          // 0..31
    const int wave  = threadIdx.x >> 5;    // 0..3 == batch tile mt
    const int lane  = threadIdx.x & 31;
    const int mt    = wave;
    const int laneN = lane & 15;
    const int khalf = lane >> 4;
    const int colBase = g32 * 16;
    const int col   = colBase + laneN;     // hidden column owned by this lane (N)
    const int brow0 = 16 * mt + 8 * khalf; // first batch row of this lane's C elems
    const int arow  = 16 * mt + laneN;     // A-fragment batch row (M)

    // ---- stage the WG's weight slices into LDS (once) ----
    // slice row r (0..63): gate g = r>>4, col c = r&15 -> global row g*512+colBase+c
    for (int id = threadIdx.x; id < 3 * 64 * 64; id += 128) {
        const int m   = id >> 12;          // matrix 0..2
        const int rem = id & 4095;
        const int r   = rem >> 6;          // slice row 0..63
        const int c8  = (rem & 63) << 3;   // k offset (multiple of 8)
        const __bf16* src = (m == 0 ? Whh0 : m == 1 ? Wih1 : Whh1)
                          + (size_t)((r >> 4) * 512 + colBase + (r & 15)) * 512 + c8;
        *(v8bf*)(smem + m * LDS_MAT + r * LDS_ROW + c8) = *(const v8bf*)src;
    }

    // Persistent register state
    v8f gxr[4], c0, c1, hacc;
    float bb[4];
#pragma unroll
    for (int g = 0; g < 4; ++g) {
        bb[g] = b1s[g * 512 + col];
#pragma unroll
        for (int r = 0; r < 8; ++r)
            gxr[g][r] = gx0[(size_t)(brow0 + r) * 2048 + g * 512 + col];
    }
#pragma unroll
    for (int r = 0; r < 8; ++r) { c0[r] = 0.f; c1[r] = 0.f; hacc[r] = 0.f; }

    __syncthreads();                       // LDS weights ready

    const __bf16* ws0 = smem;                  // Whh0 slice
    const __bf16* ws1 = smem + LDS_MAT;        // Wih1 slice
    const __bf16* ws2 = smem + 2 * LDS_MAT;    // Whh1 slice
    const int bfragOff = 16 * khalf;           // within-row k offset of B fragment

    unsigned tgt = 0;
    const unsigned nwg = gridDim.x;

    for (int t = 0; t < 1024; ++t) {
        const int pb = t & 1;
        const __bf16* Pprev = Pbuf + pb * (64 * 512);
        __bf16*       Pnext = Pbuf + (pb ^ 1) * (64 * 512);
        const __bf16* Qprev = Qbuf + pb * (64 * 512);
        __bf16*       Qnext = Qbuf + (pb ^ 1) * (64 * 512);

        // -------- layer 0: gates = gx0 + p_{t-1} @ Whh0^T --------
        v8f acc[4] = {gxr[0], gxr[1], gxr[2], gxr[3]};
        {
            const __bf16* Ab = Pprev + (size_t)arow * 512;
            for (int kc = 0; kc < 512; kc += 32) {
                v16bf a = loadA16(Ab, kc, khalf);
                __builtin_prefetch(Ab + kc + 64, 0, 3);
#pragma unroll
                for (int g = 0; g < 4; ++g) {
                    v16bf bm = loadB_lds(ws0 + (g * 16 + laneN) * LDS_ROW + kc + bfragOff);
                    acc[g] = bfwmma(a, bm, acc[g]);
                }
            }
        }
#pragma unroll
        for (int r = 0; r < 8; ++r) {
            float iv = sigmoidf_(acc[0][r]);
            float fv = sigmoidf_(acc[1][r]);
            float gv = tanhf_fast(acc[2][r]);
            float ov = sigmoidf_(acc[3][r]);
            float cv = fv * c0[r] + iv * gv;
            c0[r] = cv;
            float hv = ov * tanhf_fast(cv);
            Pnext[(size_t)(brow0 + r) * 512 + col] = f2bf(hv);
        }
        tgt += nwg;
        grid_barrier(cnt, tgt);        // p_t globally visible

        // -------- layer 1: gates = b1 + p_t @ Wih1^T + q_{t-1} @ Whh1^T --------
        v8f a1[4];
#pragma unroll
        for (int g = 0; g < 4; ++g)
#pragma unroll
            for (int r = 0; r < 8; ++r) a1[g][r] = bb[g];
        {
            const __bf16* Ap = Pnext + (size_t)arow * 512;
            const __bf16* Aq = Qprev + (size_t)arow * 512;
            for (int kc = 0; kc < 512; kc += 32) {
                v16bf av = loadA16(Ap, kc, khalf);
#pragma unroll
                for (int g = 0; g < 4; ++g) {
                    v16bf bm = loadB_lds(ws1 + (g * 16 + laneN) * LDS_ROW + kc + bfragOff);
                    a1[g] = bfwmma(av, bm, a1[g]);
                }
                v16bf aw = loadA16(Aq, kc, khalf);
#pragma unroll
                for (int g = 0; g < 4; ++g) {
                    v16bf bm = loadB_lds(ws2 + (g * 16 + laneN) * LDS_ROW + kc + bfragOff);
                    a1[g] = bfwmma(aw, bm, a1[g]);
                }
            }
        }
#pragma unroll
        for (int r = 0; r < 8; ++r) {
            float iv = sigmoidf_(a1[0][r]);
            float fv = sigmoidf_(a1[1][r]);
            float gv = tanhf_fast(a1[2][r]);
            float ov = sigmoidf_(a1[3][r]);
            float cv = fv * c1[r] + iv * gv;
            c1[r] = cv;
            float qv = ov * tanhf_fast(cv);
            __bf16 qb = f2bf(qv);
            Qnext[(size_t)(brow0 + r) * 512 + col] = qb;
            ch[((size_t)(brow0 + r) * 1024 + t) * 512 + col] = qb;
            hacc[r] += qv;
        }
        tgt += nwg;
        grid_barrier(cnt, tgt);        // q_t visible; safe to reuse buffers
    }
#pragma unroll
    for (int r = 0; r < 8; ++r)
        havg[(size_t)(brow0 + r) * 512 + col] = hacc[r] * (1.f / 1024.f);
}

// ---------------------------------------------------------------------------
// K3a/K3b: oscillator head from h_avg
// ---------------------------------------------------------------------------
__global__ void oscy_kernel(const float* __restrict__ havg, const float* __restrict__ w1,
                            const float* __restrict__ b1, float* __restrict__ yosc) {
    const int b = blockIdx.x, j = threadIdx.x;      // 64 x 256
    const float* hr = havg + (size_t)b * 512;
    const float* wr = w1 + (size_t)j * 512;
    float s = b1[j];
    for (int k = 0; k < 512; ++k) s += hr[k] * wr[k];
    yosc[b * 256 + j] = s;
}

__global__ void oscp_kernel(const float* __restrict__ yosc, const float* __restrict__ g,
                            const float* __restrict__ be, const float* __restrict__ w2,
                            const float* __restrict__ b2, float* __restrict__ oscp) {
    const int b = threadIdx.x;
    if (b >= 64) return;
    const float* yr = yosc + b * 256;
    float m = 0.f;
    for (int j = 0; j < 256; ++j) m += yr[j];
    m *= (1.f / 256.f);
    float v = 0.f;
    for (int j = 0; j < 256; ++j) { float d = yr[j] - m; v += d * d; }
    v *= (1.f / 256.f);
    const float inv = rsqrtf(v + 1e-5f);
    float op0 = b2[0], op1 = b2[1], op2 = b2[2];
    for (int j = 0; j < 256; ++j) {
        float xn = (yr[j] - m) * inv * g[j] + be[j];
        xn = xn >= 0.f ? xn : 0.2f * xn;
        op0 += xn * w2[0 * 256 + j];
        op1 += xn * w2[1 * 256 + j];
        op2 += xn * w2[2 * 256 + j];
    }
    oscp[b * 4 + 0] = 0.23f + 0.04f * tanhf(op0);
    oscp[b * 4 + 1] = 2.0f + 1.5f * tanhf(op1);
    oscp[b * 4 + 2] = 3.14159265358979323846f * sigmoidf_(op2);
}

// ---------------------------------------------------------------------------
// K4: sig head — base[b,t] = tanh(lrelu(ln(ch@W1^T+b1)) . w2 + b2), WMMA GEMM
//  One WG per 16 flattened (b,t) rows; 8 waves x 2 N-tiles = 256 cols.
// ---------------------------------------------------------------------------
__global__ __launch_bounds__(256) void sighead_kernel(
    const __bf16* __restrict__ ch, const __bf16* __restrict__ W1,
    const float* __restrict__ b1, const float* __restrict__ g1,
    const float* __restrict__ be1, const float* __restrict__ w2,
    const float* __restrict__ b2, float* __restrict__ base) {
    __shared__ float y[16 * 256];
    const int wave = threadIdx.x >> 5, lane = threadIdx.x & 31;
    const int laneN = lane & 15, khalf = lane >> 4;
    const int R0 = blockIdx.x * 16;
    const __bf16* Ab = ch + (size_t)(R0 + laneN) * 512;
#pragma unroll
    for (int s = 0; s < 2; ++s) {
        const int ncol = (wave * 2 + s) * 16 + laneN;
        v8f acc;
        const float bv = b1[ncol];
#pragma unroll
        for (int r = 0; r < 8; ++r) acc[r] = bv;
        const __bf16* Brow = W1 + (size_t)ncol * 512;
        for (int kc = 0; kc < 512; kc += 32) {
            v16bf a = loadA16(Ab, kc, khalf);
            // B fragment: K = kc + 16*khalf .. +16 contiguous, 32B-aligned
            v16bf bm = *(const v16bf*)(Brow + kc + 16 * khalf);
            acc = bfwmma(a, bm, acc);
        }
#pragma unroll
        for (int r = 0; r < 8; ++r)
            y[(r + 8 * khalf) * 256 + ncol] = acc[r];
    }
    __syncthreads();
    if (threadIdx.x < 16) {
        const int r = threadIdx.x;
        float m = 0.f;
        for (int j = 0; j < 256; ++j) m += y[r * 256 + j];
        m *= (1.f / 256.f);
        float v = 0.f;
        for (int j = 0; j < 256; ++j) { float d = y[r * 256 + j] - m; v += d * d; }
        v *= (1.f / 256.f);
        const float inv = rsqrtf(v + 1e-5f);
        float s = 0.f;
        for (int j = 0; j < 256; ++j) {
            float xn = (y[r * 256 + j] - m) * inv * g1[j] + be1[j];
            xn = xn >= 0.f ? xn : 0.2f * xn;
            s += xn * w2[j];
        }
        base[R0 + r] = tanhf(s + b2[0]);
    }
}

// ---------------------------------------------------------------------------
// K5: enh/stress/smooth/label-select -> out (64,1024,1)
// ---------------------------------------------------------------------------
__global__ void final_kernel(const float* __restrict__ base, const float* __restrict__ oscp,
                             const int* __restrict__ labels, const float* __restrict__ sw,
                             const float* __restrict__ sb, const float* __restrict__ aw,
                             const float* __restrict__ ab, float* __restrict__ out) {
    const int idx = blockIdx.x * blockDim.x + threadIdx.x;   // 65536
    const int b = idx >> 10, t = idx & 1023;
    const float fr = oscp[b * 4 + 0], am = oscp[b * 4 + 1], ph = oscp[b * 4 + 2];
    const float TWO_PI = 6.283185307179586476925f;
    auto eAt = [&](int tt) -> float {
        float ti = (float)tt * (1.0f / 1023.0f);
        float osc = am * sinf(TWO_PI * fr * 1024.0f * ti + ph);
        return 0.6f * base[b * 1024 + tt] + 0.4f * osc;
    };
    const float e  = eAt(t);
    const float em = (t > 0)    ? eAt(t - 1) : 0.f;
    const float ep = (t < 1023) ? eAt(t + 1) : 0.f;
    const int lab = labels[b];
    const float stress = sw[0] * e + sb[0];
    const float smooth = aw[0] * em + aw[1] * e + aw[2] * ep + ab[0];
    out[idx] = (lab == 1) ? e : (lab == 2) ? stress : (lab == 3) ? smooth : 0.f;
}

// ---------------------------------------------------------------------------
// kernel_launch
// ---------------------------------------------------------------------------
extern "C" void kernel_launch(void* const* d_in, const int* in_sizes, int n_in,
                              void* d_out, int out_size, void* d_ws, size_t ws_size,
                              hipStream_t stream) {
    const float* z      = (const float*)d_in[0];
    const int*   labels = (const int*)  d_in[1];
    const float* emb    = (const float*)d_in[2];
    const float* npw    = (const float*)d_in[3];
    const float* npb    = (const float*)d_in[4];
    const float* npg    = (const float*)d_in[5];
    const float* npbeta = (const float*)d_in[6];
    const float* wih0   = (const float*)d_in[7];
    const float* whh0   = (const float*)d_in[8];
    const float* bih0   = (const float*)d_in[9];
    const float* bhh0   = (const float*)d_in[10];
    const float* wih1   = (const float*)d_in[11];
    const float* whh1   = (const float*)d_in[12];
    const float* bih1   = (const float*)d_in[13];
    const float* bhh1   = (const float*)d_in[14];
    const float* oscw1  = (const float*)d_in[15];
    const float* oscb1  = (const float*)d_in[16];
    const float* oscg   = (const float*)d_in[17];
    const float* oscbe  = (const float*)d_in[18];
    const float* oscw2  = (const float*)d_in[19];
    const float* oscb2  = (const float*)d_in[20];
    const float* sigw1  = (const float*)d_in[21];
    const float* sigb1  = (const float*)d_in[22];
    const float* sigg   = (const float*)d_in[23];
    const float* sigbe  = (const float*)d_in[24];
    const float* sigw2  = (const float*)d_in[25];
    const float* sigb2  = (const float*)d_in[26];
    const float* stw    = (const float*)d_in[27];
    const float* stb    = (const float*)d_in[28];
    const float* amw    = (const float*)d_in[29];
    const float* amb    = (const float*)d_in[30];

    char* w = (char*)d_ws;
    unsigned* cnt   = (unsigned*)(w + WS_CNT);
    __bf16* Pb      = (__bf16*)(w + WS_PBUF);
    __bf16* Qb      = (__bf16*)(w + WS_QBUF);
    float*  xin2    = (float*) (w + WS_XIN2);
    float*  gx0     = (float*) (w + WS_GX0);
    __bf16* Whh0b   = (__bf16*)(w + WS_WHH0);
    __bf16* Wih1b   = (__bf16*)(w + WS_WIH1);
    __bf16* Whh1b   = (__bf16*)(w + WS_WHH1);
    __bf16* W1sb    = (__bf16*)(w + WS_W1S);
    float*  b1s     = (float*) (w + WS_B1S);
    __bf16* chb     = (__bf16*)(w + WS_CH);
    float*  havg    = (float*) (w + WS_HAVG);
    float*  yosc    = (float*) (w + WS_YOSC);
    float*  oscp    = (float*) (w + WS_OSCP);
    float*  baseb   = (float*) (w + WS_BASE);
    float*  out     = (float*) d_out;

    // allow >64KB dynamic LDS for the scan kernel (WGP has 320KB)
    (void)hipFuncSetAttribute((const void*)lstm_scan_kernel,
                              hipFuncAttributeMaxDynamicSharedMemorySize, LDS_BYTES);

    // K0: zero counter + both hidden-state double buffers (262400 B = 65600 u32)
    zero_kernel<<<257, 256, 0, stream>>>((unsigned*)w, 65600);
    // K1: prep head, time-invariant gate precompute, bf16 weight repack
    prep_kernel<<<64, 256, 0, stream>>>(z, labels, emb, npw, npb, npg, npbeta, xin2);
    gx0_kernel<<<512, 256, 0, stream>>>(xin2, wih0, bih0, bhh0, gx0);
    cvt_kernel<<<4096, 256, 0, stream>>>(whh0, Whh0b, 2048 * 512);
    cvt_kernel<<<4096, 256, 0, stream>>>(wih1, Wih1b, 2048 * 512);
    cvt_kernel<<<4096, 256, 0, stream>>>(whh1, Whh1b, 2048 * 512);
    cvt_kernel<<<512, 256, 0, stream>>>(sigw1, W1sb, 256 * 512);
    bsum_kernel<<<8, 256, 0, stream>>>(bih1, bhh1, b1s, 2048);
    // K2: fused 2-layer LSTM scan (WMMA + LDS-resident weights, 32 coop WGs)
    lstm_scan_kernel<<<32, 128, LDS_BYTES, stream>>>(gx0, Whh0b, Wih1b, Whh1b, b1s,
                                                     Pb, Qb, chb, havg, cnt);
    // K3: oscillator head
    oscy_kernel<<<64, 256, 0, stream>>>(havg, oscw1, oscb1, yosc);
    oscp_kernel<<<1, 64, 0, stream>>>(yosc, oscg, oscbe, oscw2, oscb2, oscp);
    // K4: sig head GEMM (WMMA) + per-row LN + projection
    sighead_kernel<<<4096, 256, 0, stream>>>(chb, W1sb, sigb1, sigg, sigbe,
                                             sigw2, sigb2, baseb);
    // K5: combine + label select
    final_kernel<<<256, 256, 0, stream>>>(baseb, oscp, labels, stw, stb, amw, amb, out);
}
